// SparseLogicBlock_72859825209821
// MI455X (gfx1250) — 8-bit matrix operand order repaired and checked by reading, compiled, not run on hardware
//
#include <hip/hip_runtime.h>
#include <math.h>

#define EPSF 1e-5f

typedef __attribute__((ext_vector_type(8))) int v8i;

union FragI {
  v8i v;
  int i[8];
  unsigned u[8];
  uint2 d2[4];
  uint4 d4[2];
};

static __device__ __forceinline__ signed char quant_i8(float f) {
  return (signed char)fminf(fmaxf(rintf(f), -128.f), 127.f);
}

static __device__ __forceinline__ int pk_byte(int k) {
  return (k & ~63) | ((k & 16) << 1) | ((k & 32) >> 1) | (k & 15);
}

static __device__ __forceinline__ void async_copy16(unsigned lds_dst,
                                                    const signed char* g) {
  asm volatile("global_load_async_to_lds_b128 %0, %1, off"
               :
               : "v"(lds_dst), "v"(g)
               : "memory");
}
static __device__ __forceinline__ void async_wait0() {
  asm volatile("s_wait_asynccnt 0" ::: "memory");
}

__global__ __launch_bounds__(256) void abs_sum_kernel(
    const float* __restrict__ W, long long n4, float* __restrict__ out) {
  __shared__ float red[256];
  long long i = (long long)blockIdx.x * blockDim.x + threadIdx.x;
  const long long stride = (long long)gridDim.x * blockDim.x;
  float s = 0.f;
  const float4* W4 = (const float4*)W;
  for (; i < n4; i += stride) {
    float4 v = W4[i];
    s += fabsf(v.x) + fabsf(v.y) + fabsf(v.z) + fabsf(v.w);
  }
  red[threadIdx.x] = s;
  __syncthreads();
  for (int o = 128; o > 0; o >>= 1) {
    if (threadIdx.x < o) red[threadIdx.x] += red[threadIdx.x + o];
    __syncthreads();
  }
  if (threadIdx.x == 0) atomicAdd(out, red[0]);
}

__global__ __launch_bounds__(256) void quant_weight_kernel(
    const float* __restrict__ W, signed char* __restrict__ Qp, int O, int K,
    const float* __restrict__ wsum, float rcount) {
  const float ws = wsum[0] * rcount + EPSF;
  const float rws = 1.f / ws;
  const int o = blockIdx.x * 256 + threadIdx.x;
  const int kq = blockIdx.y;
  const float4 wv = *(const float4*)(W + (size_t)o * K + (size_t)kq * 4);
  char4 q;
  q.x = (signed char)fminf(fmaxf(rintf(wv.x * rws), -1.f), 1.f);
  q.y = (signed char)fminf(fmaxf(rintf(wv.y * rws), -1.f), 1.f);
  q.z = (signed char)fminf(fmaxf(rintf(wv.z * rws), -1.f), 1.f);
  q.w = (signed char)fminf(fmaxf(rintf(wv.w * rws), -1.f), 1.f);
  ((char4*)Qp)[((size_t)(kq >> 2) * O + o) * 4 + (kq & 3)] = q;
}

template <int MODE>
__global__ __launch_bounds__(256) void ln_kernel(
    const float* __restrict__ X, const float* __restrict__ w,
    const float* __restrict__ bia, float* __restrict__ Y,
    float* __restrict__ rowmax, signed char* __restrict__ Q,
    float* __restrict__ deq) {
  __shared__ float r1[256], r2[256];
  const int row = blockIdx.x;
  const int t = threadIdx.x;
  const float4 v = ((const float4*)X)[(size_t)row * 256 + t];
  r1[t] = v.x + v.y + v.z + v.w;
  r2[t] = v.x * v.x + v.y * v.y + v.z * v.z + v.w * v.w;
  __syncthreads();
  for (int o = 128; o > 0; o >>= 1) {
    if (t < o) { r1[t] += r1[t + o]; r2[t] += r2[t + o]; }
    __syncthreads();
  }
  const float mean = r1[0] * (1.f / 1024.f);
  const float var = r2[0] * (1.f / 1024.f) - mean * mean;
  __syncthreads();
  const float rs = rsqrtf(var + EPSF);
  const float4 wv = ((const float4*)w)[t];
  const float4 bv = ((const float4*)bia)[t];
  float4 y;
  y.x = (v.x - mean) * rs * wv.x + bv.x;
  y.y = (v.y - mean) * rs * wv.y + bv.y;
  y.z = (v.z - mean) * rs * wv.z + bv.z;
  y.w = (v.w - mean) * rs * wv.w + bv.w;
  float am = fmaxf(fmaxf(fabsf(y.x), fabsf(y.y)), fmaxf(fabsf(y.z), fabsf(y.w)));
  r1[t] = am;
  __syncthreads();
  for (int o = 128; o > 0; o >>= 1) {
    if (t < o) r1[t] = fmaxf(r1[t], r1[t + o]);
    __syncthreads();
  }
  const float M = r1[0];
  if (MODE == 0) {
    ((float4*)Y)[(size_t)row * 256 + t] = y;
    if (t == 0) rowmax[row] = M;
  } else {
    const float Mc = fmaxf(M, EPSF);
    const float xs = 127.f / Mc;
    char4 q;
    q.x = quant_i8(y.x * xs);
    q.y = quant_i8(y.y * xs);
    q.z = quant_i8(y.z * xs);
    q.w = quant_i8(y.w * xs);
    ((char4*)Q)[(size_t)row * 256 + (pk_byte(4 * t) >> 2)] = q;
    if (t == 0) deq[row] = Mc * (1.f / 127.f);
  }
}

__global__ __launch_bounds__(256) void build_combined_kernel(
    const float* __restrict__ xn, const float* __restrict__ rowmax,
    const int* __restrict__ s0, const int* __restrict__ s1,
    const int* __restrict__ iv, signed char* __restrict__ cq,
    float* __restrict__ deqA) {
  __shared__ int srcs[15];
  __shared__ float sxs;
  const int row = blockIdx.x;
  const int b = row >> 8;
  const int l = row & 255;
  const int t = threadIdx.x;
  if (t == 0) srcs[0] = l;
  else if (t < 4) srcs[t] = s0[l * 3 + (t - 1)];
  else if (t < 7) srcs[t] = s1[l * 3 + (t - 4)];
  else if (t < 15) srcs[t] = iv[l * 8 + (t - 7)];
  __syncthreads();
  if (t == 0) {
    float m = 0.f;
    for (int i = 0; i < 15; ++i) m = fmaxf(m, rowmax[b * 256 + srcs[i]]);
    m = fmaxf(m, EPSF);
    sxs = 127.f / m;
    deqA[row] = m * (1.f / 127.f);
  }
  __syncthreads();
  const float xs = sxs;
  const int pidx = pk_byte(4 * t) >> 2;
#pragma unroll
  for (int seg = 0; seg < 15; ++seg) {
    const float4 v = ((const float4*)xn)[(size_t)(b * 256 + srcs[seg]) * 256 + t];
    char4 q;
    q.x = quant_i8(v.x * xs);
    q.y = quant_i8(v.y * xs);
    q.z = quant_i8(v.z * xs);
    q.w = quant_i8(v.w * xs);
    ((char4*)cq)[(size_t)row * 3840 + seg * 256 + pidx] = q;
  }
}

__global__ __launch_bounds__(256) void quant_rows_kernel(
    const float* __restrict__ X, signed char* __restrict__ Q,
    float* __restrict__ deq, int n4) {
  __shared__ float r1[256];
  const int row = blockIdx.x;
  const int t = threadIdx.x;
  const int iters = n4 >> 8;
  float4 vv[4];
  float am = 0.f;
#pragma unroll 4
  for (int i = 0; i < iters; ++i) {
    vv[i] = ((const float4*)X)[(size_t)row * n4 + i * 256 + t];
    am = fmaxf(am, fmaxf(fmaxf(fabsf(vv[i].x), fabsf(vv[i].y)),
                         fmaxf(fabsf(vv[i].z), fabsf(vv[i].w))));
  }
  r1[t] = am;
  __syncthreads();
  for (int o = 128; o > 0; o >>= 1) {
    if (t < o) r1[t] = fmaxf(r1[t], r1[t + o]);
    __syncthreads();
  }
  const float M = fmaxf(r1[0], EPSF);
  const float xs = 127.f / M;
#pragma unroll 4
  for (int i = 0; i < iters; ++i) {
    char4 q;
    q.x = quant_i8(vv[i].x * xs);
    q.y = quant_i8(vv[i].y * xs);
    q.z = quant_i8(vv[i].z * xs);
    q.w = quant_i8(vv[i].w * xs);
    ((char4*)Q)[(size_t)row * n4 + (pk_byte((i * 256 + t) * 4) >> 2)] = q;
  }
  if (t == 0) deq[row] = M * (1.f / 127.f);
}

template <int MODE>
__global__ __launch_bounds__(256) void gemm_i8_kernel(
    const signed char* __restrict__ A, const signed char* __restrict__ Bp,
    int N, int K, const float* __restrict__ wsum, float rcount,
    const float* __restrict__ deq, const float* __restrict__ resid,
    float* __restrict__ out) {
  __shared__ __align__(16) unsigned char sAb[2][128 * 96];
  __shared__ __align__(16) unsigned char sBb[2][4 * 2048];

  const int t = threadIdx.x;
  const int lane = t & 31;
  const int wid = t >> 5;
  const int wm = wid >> 2;
  const int wn = wid & 3;
  const int half = lane >> 4;
  const int ml = lane & 15;

  const int row0 = blockIdx.y * 128;
  const int col0 = blockIdx.x * 128;

  const float wscale = wsum[0] * rcount + EPSF;

  const int ar = t >> 1;
  const int ac = (t & 1) << 5;
  const int bkg = t >> 6;
  const int bc = (t & 63) << 5;
  const signed char* gA = A + (size_t)(row0 + ar) * K + ac;
  const signed char* gB =
      Bp + (size_t)bkg * ((size_t)N * 16) + (size_t)col0 * 16 + bc;
  const size_t gBstep = (size_t)N * 64;

  const unsigned ldsA0 = (unsigned)(size_t)(void*)sAb[0];
  const unsigned ldsA1 = (unsigned)(size_t)(void*)sAb[1];
  const unsigned ldsB0 = (unsigned)(size_t)(void*)sBb[0];
  const unsigned ldsB1 = (unsigned)(size_t)(void*)sBb[1];
  const unsigned aoff = (unsigned)(ar * 96 + ac);
  const unsigned boff = (unsigned)(bkg * 2048 + bc);

  FragI acc[4][2];
#pragma unroll
  for (int mt = 0; mt < 4; ++mt)
#pragma unroll
    for (int nt = 0; nt < 2; ++nt)
#pragma unroll
      for (int r = 0; r < 8; ++r) acc[mt][nt].i[r] = 0;

  const int KS = K >> 6;

  async_copy16(ldsA0 + aoff, gA);
  async_copy16(ldsA0 + aoff + 16, gA + 16);
  async_copy16(ldsB0 + boff, gB);
  async_copy16(ldsB0 + boff + 16, gB + 16);
  async_wait0();
  __syncthreads();

  for (int ks = 0; ks < KS; ++ks) {
    const int buf = ks & 1;
    if (ks + 1 < KS) {
      const signed char* nA = gA + (size_t)(ks + 1) * 64;
      const signed char* nB = gB + (size_t)(ks + 1) * gBstep;
      const unsigned la = (buf ? ldsA0 : ldsA1) + aoff;
      const unsigned lb = (buf ? ldsB0 : ldsB1) + boff;
      async_copy16(la, nA);
      async_copy16(la + 16, nA + 16);
      async_copy16(lb, nB);
      async_copy16(lb + 16, nB + 16);
    }

    const unsigned char* sa = (const unsigned char*)sAb[buf];
    const unsigned char* sb = (const unsigned char*)sBb[buf];

    FragI fa[4], fb[2];
#pragma unroll
    for (int mt = 0; mt < 4; ++mt) {
      const unsigned char* p = sa + (wm * 64 + mt * 16 + ml) * 96 + (half << 5);
      fa[mt].d4[0] = *(const uint4*)(p);
      fa[mt].d4[1] = *(const uint4*)(p + 16);
    }
#pragma unroll
    for (int nt = 0; nt < 2; ++nt) {
      const unsigned char* p = sb + (wn * 32 + nt * 16 + ml) * 16;
      fb[nt].d4[0] = *(const uint4*)(p + half * 2048);
      fb[nt].d4[1] = *(const uint4*)(p + (2 + half) * 2048);
    }
#pragma unroll
    for (int mt = 0; mt < 4; ++mt)
#pragma unroll
      for (int nt = 0; nt < 2; ++nt)
        acc[mt][nt].v = __builtin_amdgcn_wmma_i32_16x16x64_iu8(
            true, fa[mt].v, true, fb[nt].v, acc[mt][nt].v, false, false);

    if (ks + 1 < KS) {
      async_wait0();
      __syncthreads();
    }
  }

#pragma unroll
  for (int mt = 0; mt < 4; ++mt) {
#pragma unroll
    for (int nt = 0; nt < 2; ++nt) {
      const int gcol = col0 + wn * 32 + nt * 16 + ml;
#pragma unroll
      for (int r = 0; r < 8; ++r) {
        const int grow = row0 + wm * 64 + mt * 16 + (half << 3) + r;
        float f = (float)acc[mt][nt].i[r] * wscale * deq[grow];
        const size_t oidx = (size_t)grow * N + gcol;
        if (MODE == 0 || MODE == 2) f += resid[oidx];
        if (MODE == 1) f = 0.5f * f * (1.f + erff(f * 0.70710678118654752f));
        out[oidx] = f;
      }
    }
  }
}

extern "C" void kernel_launch(void* const* d_in, const int* in_sizes, int n_in,
                              void* d_out, int out_size, void* d_ws,
                              size_t ws_size, hipStream_t stream) {
  (void)in_sizes; (void)n_in; (void)out_size; (void)ws_size;
  const float* x = (const float*)d_in[0];
  const float* ln1w = (const float*)d_in[1];
  const float* ln1b = (const float*)d_in[2];
  const float* ln2w = (const float*)d_in[3];
  const float* ln2b = (const float*)d_in[4];
  const float* Wg = (const float*)d_in[5];
  const float* W1 = (const float*)d_in[6];
  const float* W2 = (const float*)d_in[7];
  const int* idx_s0 = (const int*)d_in[8];
  const int* idx_s1 = (const int*)d_in[9];
  const int* idx_v = (const int*)d_in[10];

  const int D = 1024, BL = 8192;
  const int K1 = 15 * D, N1 = D;
  const int K2 = D, N2 = 4 * D;
  const int K3 = 4 * D, N3 = D;

  char* wsb = (char*)d_ws;
  size_t off = 0;
  auto bump = [&](size_t sz) -> char* {
    char* p = wsb + off;
    off += (sz + 255) & ~(size_t)255;
    return p;
  };
  float* sums = (float*)bump(256);
  float* xn = (float*)bump((size_t)BL * D * 4);
  float* x2 = (float*)bump((size_t)BL * D * 4);
  char* reg1 = bump((size_t)BL * N2 * 4);
  signed char* Wgq = (signed char*)bump((size_t)K1 * N1);
  signed char* W1q = (signed char*)bump((size_t)K2 * N2);
  signed char* W2q = (signed char*)bump((size_t)K3 * N3);
  signed char* h8 = (signed char*)bump((size_t)BL * D);
  signed char* gq = (signed char*)bump((size_t)BL * N2);
  float* deqA = (float*)bump((size_t)BL * 4);
  float* deqH = (float*)bump((size_t)BL * 4);
  float* deqG = (float*)bump((size_t)BL * 4);
  float* rmax = (float*)bump((size_t)BL * 4);
  signed char* cq = (signed char*)reg1;
  float* gbuf = (float*)reg1;

  const float rc1 = 1.f / ((float)N1 * (float)K1);
  const float rc2 = 1.f / ((float)N2 * (float)K2);
  const float rc3 = 1.f / ((float)N3 * (float)K3);

  hipMemsetAsync(sums, 0, 256, stream);
  abs_sum_kernel<<<512, 256, 0, stream>>>(Wg, (long long)N1 * K1 / 4, &sums[0]);
  abs_sum_kernel<<<512, 256, 0, stream>>>(W1, (long long)N2 * K2 / 4, &sums[1]);
  abs_sum_kernel<<<512, 256, 0, stream>>>(W2, (long long)N3 * K3 / 4, &sums[2]);

  quant_weight_kernel<<<dim3(N1 / 256, K1 / 4), 256, 0, stream>>>(
      Wg, Wgq, N1, K1, &sums[0], rc1);
  quant_weight_kernel<<<dim3(N2 / 256, K2 / 4), 256, 0, stream>>>(
      W1, W1q, N2, K2, &sums[1], rc2);
  quant_weight_kernel<<<dim3(N3 / 256, K3 / 4), 256, 0, stream>>>(
      W2, W2q, N3, K3, &sums[2], rc3);

  ln_kernel<0><<<BL, 256, 0, stream>>>(x, ln1w, ln1b, xn, rmax, nullptr,
                                       nullptr);
  build_combined_kernel<<<BL, 256, 0, stream>>>(xn, rmax, idx_s0, idx_s1,
                                                idx_v, cq, deqA);
  gemm_i8_kernel<0><<<dim3(N1 / 128, BL / 128), 256, 0, stream>>>(
      cq, Wgq, N1, K1, &sums[0], rc1, deqA, x, x2);

  ln_kernel<1><<<BL, 256, 0, stream>>>(x2, ln2w, ln2b, nullptr, nullptr, h8,
                                       deqH);
  gemm_i8_kernel<1><<<dim3(N2 / 128, BL / 128), 256, 0, stream>>>(
      h8, W1q, N2, K2, &sums[1], rc2, deqH, nullptr, gbuf);
  quant_rows_kernel<<<BL, 256, 0, stream>>>(gbuf, gq, deqG, N2 / 4);
  gemm_i8_kernel<2><<<dim3(N3 / 128, BL / 128), 256, 0, stream>>>(
      gq, W2q, N3, K3, &sums[2], rc3, deqG, x2, (float*)d_out);
}
